// GraphClassifier_8667244003518
// MI455X (gfx1250) — matrix operand order repaired and checked by reading, not compiled, not run on hardware
//
#include <hip/hip_runtime.h>
#include <math.h>

#define HDIM 128
#define GNUM 512
#define EPSC 1e-5f

typedef __attribute__((ext_vector_type(16))) _Float16 v16h;
typedef __attribute__((ext_vector_type(8)))  float    v8f;

// ============================================================================
// WMMA GEMM:  Out[rows x co] = epi( (A (+A2))[rows x K] @ W[K x co] + bias )
//   epi 0: acc+bias        epi 1: relu(acc+bias)
//   epi 2: relu(res + relu(acc+bias))      (GINE residual update, in-place OK)
// Requires K % 32 == 0 (host pads otherwise), co % 16 == 0.
//
// Block = 128 threads = 4 waves; block owns a 64-row x 16-col output strip.
// The K x 16 weight strip is staged in LDS once and shared by all 4 waves
// across the whole K loop (ds_load). A is read directly from global as
// branch-free float2 (rows clamped, stores masked); x/agg/h1 are L2-resident
// (192 MB L2), so the co/16-fold A re-reads never reach HBM.
// Fragment addressing per CDNA5 ISA 7.12.2 (wave32):
//   A 16x32 f16 : lane L -> row M = L&15, g = L>>4;
//                 pair j : K = (j<4 ? 8g+2j : 16+8g+2(j-4)) , +1
//   B 32x16 f16 : lane L -> col N = L&15; pair j: K = (j<4 ? 8g+2j : 16+8g+2(j-4)) , +1
//   C/D 16x16 f32: lane L -> col N = L&15; elem j -> row M = j + 8g
// ============================================================================
template <bool HASA2>
__global__ void wmma_gemm_t(const float* __restrict__ A,
                            const float* __restrict__ A2,
                            const float* __restrict__ W,
                            const float* __restrict__ bias,
                            const float* __restrict__ res,
                            float* __restrict__ Out,
                            int rows, int K, int co, int epi)
{
  __shared__ float ldsW[256 * 16];                 // max K = 256 -> 16 KB

  const int tid  = threadIdx.x;
  const int lane = tid & 31;
  const int wave = tid >> 5;
  const int colTiles = co >> 4;
  const int colTile  = blockIdx.x % colTiles;
  const int rowBlk   = blockIdx.x / colTiles;
  const int col0 = colTile << 4;

  // ---- stage W[:, col0:col0+16] into LDS ----------------------------------
  for (int idx = tid; idx < K * 16; idx += 128) {
    const int kr = idx >> 4;
    const int cc = idx & 15;
    ldsW[idx] = W[(size_t)kr * co + col0 + cc];
  }
  __syncthreads();

  const int row0 = (rowBlk * 4 + wave) << 4;
  if (row0 >= rows) return;                        // wave-uniform, after barrier

  const int g  = lane >> 4;
  const int mr = lane & 15;
  int arow = row0 + mr;
  if (arow >= rows) arow = rows - 1;               // clamp: store is masked later
  const float* Ap  = A + (size_t)arow * K;
  const float* A2p = HASA2 ? (A2 + (size_t)arow * K) : nullptr;

  v8f acc = {};
  for (int kb = 0; kb < K; kb += 32) {
    v16h a, b;
#pragma unroll
    for (int j = 0; j < 8; ++j) {
      const int kk = kb + ((j < 4) ? (8 * g + 2 * j) : (16 + 8 * g + 2 * (j - 4)));
      const float2 av = *(const float2*)(Ap + kk);
      float f0 = av.x, f1 = av.y;
      if (HASA2) {
        const float2 a2 = *(const float2*)(A2p + kk);
        f0 += a2.x; f1 += a2.y;
      }
      a[2 * j]     = (_Float16)f0;
      a[2 * j + 1] = (_Float16)f1;
    }
#pragma unroll
    for (int j = 0; j < 8; ++j) {
      const int kk = kb + ((j < 4) ? (8 * g + 2 * j) : (16 + 8 * g + 2 * (j - 4)));
      b[2 * j]     = (_Float16)ldsW[kk * 16 + mr];
      b[2 * j + 1] = (_Float16)ldsW[(kk + 1) * 16 + mr];
    }
    acc = __builtin_amdgcn_wmma_f32_16x16x32_f16(false, a, false, b,
                                                 (short)0, acc, false, false);
  }

  const float bv = bias[col0 + mr];
#pragma unroll
  for (int j = 0; j < 8; ++j) {
    const int r = row0 + j + 8 * g;
    if (r < rows) {
      float v = acc[j] + bv;
      if (epi >= 1) v = fmaxf(v, 0.f);
      if (epi == 2) v = fmaxf(res[(size_t)r * co + col0 + mr] + v, 0.f);
      Out[(size_t)r * co + col0 + mr] = v;
    }
  }
}

// Zero-pad rows: Out[r, 0:Kin) = In[r, 0:Kin), Out[r, Kin:Kout) = 0
__global__ void pad_rows_kernel(const float* __restrict__ In, float* __restrict__ Out,
                                int rows, int Kin, int Kout)
{
  long t = (long)blockIdx.x * blockDim.x + threadIdx.x;
  if (t >= (long)rows * Kout) return;
  int r = (int)(t / Kout);
  int c = (int)(t % Kout);
  Out[t] = (c < Kin) ? In[(size_t)r * Kin + c] : 0.f;
}

// ============================================================================
// msg = relu(x[src] + e); agg[dst] += msg.   One thread per (edge, 4 features).
// ============================================================================
__global__ void msg_agg_kernel(const float* __restrict__ x,
                               const float* __restrict__ e,
                               const int* __restrict__ src,
                               const int* __restrict__ dst,
                               float* __restrict__ agg, int E_)
{
  int t = blockIdx.x * blockDim.x + threadIdx.x;
  if (t >= E_ * (HDIM / 4)) return;
  int edge = t >> 5;            // HDIM/4 == 32 chunks per edge
  int f = (t & 31) << 2;
  float4 xv = *(const float4*)(x + (size_t)src[edge] * HDIM + f);
  float4 ev = *(const float4*)(e + (size_t)edge * HDIM + f);
  float m0 = fmaxf(xv.x + ev.x, 0.f), m1 = fmaxf(xv.y + ev.y, 0.f);
  float m2 = fmaxf(xv.z + ev.z, 0.f), m3 = fmaxf(xv.w + ev.w, 0.f);
  float* ap = agg + (size_t)dst[edge] * HDIM + f;
  atomicAdd(ap + 0, m0); atomicAdd(ap + 1, m1);
  atomicAdd(ap + 2, m2); atomicAdd(ap + 3, m3);
}

// ============================================================================
// BatchNorm column stats: threads cover columns (coalesced), blocks cover rows.
// ============================================================================
__global__ void col_stats_kernel(const float* __restrict__ Hm,
                                 float* __restrict__ sum, float* __restrict__ sq,
                                 int rows, int cols, int rowsPer)
{
  int c = threadIdx.x;                 // blockDim.x == cols
  int r0 = blockIdx.x * rowsPer;
  int r1 = r0 + rowsPer; if (r1 > rows) r1 = rows;
  float s = 0.f, q = 0.f;
  for (int r = r0; r < r1; ++r) {
    float v = Hm[(size_t)r * cols + c];
    s += v; q += v * v;
  }
  atomicAdd(&sum[c], s);
  atomicAdd(&sq[c], q);
}

__global__ void bn_relu_kernel(float* __restrict__ Hm,
                               const float* __restrict__ sum, const float* __restrict__ sq,
                               const float* __restrict__ gamma, const float* __restrict__ beta,
                               int rows, int cols)
{
  long t = (long)blockIdx.x * blockDim.x + threadIdx.x;
  if (t >= (long)rows * cols) return;
  int c = (int)(t % cols);
  float inv = 1.f / (float)rows;
  float mean = sum[c] * inv;
  float var  = sq[c] * inv - mean * mean;
  float sc = gamma[c] * rsqrtf(var + EPSC);
  Hm[t] = fmaxf((Hm[t] - mean) * sc + beta[c], 0.f);
}

// ============================================================================
// Graph LayerNorm: per-node row sum / sumsq via wave32 shuffle reduce,
// atomically accumulated per graph.
// ============================================================================
__global__ void ln_stats_kernel(const float* __restrict__ x, const int* __restrict__ batch,
                                float* __restrict__ gsum, float* __restrict__ gsq, int rows)
{
  int node = blockIdx.x * (blockDim.x >> 5) + (threadIdx.x >> 5);
  if (node >= rows) return;
  int lane = threadIdx.x & 31;
  float4 v = *(const float4*)(x + (size_t)node * HDIM + lane * 4);
  float s = v.x + v.y + v.z + v.w;
  float q = v.x * v.x + v.y * v.y + v.z * v.z + v.w * v.w;
  for (int off = 16; off > 0; off >>= 1) {
    s += __shfl_down(s, off, 32);
    q += __shfl_down(q, off, 32);
  }
  if (lane == 0) {
    int g = batch[node];
    atomicAdd(&gsum[g], s);
    atomicAdd(&gsq[g], q);
  }
}

__global__ void ln_apply_kernel(float* __restrict__ x, const int* __restrict__ batch,
                                const float* __restrict__ gsum, const float* __restrict__ gsq,
                                const float* __restrict__ cnt,
                                const float* __restrict__ gamma, const float* __restrict__ beta,
                                int rows)
{
  long t = (long)blockIdx.x * blockDim.x + threadIdx.x;
  if (t >= (long)rows * HDIM) return;
  int node = (int)(t / HDIM);
  int c = (int)(t % HDIM);
  int g = batch[node];
  float denom = cnt[g] * (float)HDIM;
  float mean = gsum[g] / denom;
  float var  = gsq[g] / denom - mean * mean;
  x[t] = gamma[c] * (x[t] - mean) * rsqrtf(var + EPSC) + beta[c];
}

__global__ void count_nodes_kernel(const int* __restrict__ batch, float* __restrict__ cnt, int rows)
{
  int n = blockIdx.x * blockDim.x + threadIdx.x;
  if (n < rows) atomicAdd(&cnt[batch[n]], 1.0f);
}

// ============================================================================
// Softmax-attention pooling helpers
// ============================================================================
__device__ __forceinline__ unsigned f2ord(float f) {
  unsigned u = __float_as_uint(f);
  return (u & 0x80000000u) ? ~u : (u | 0x80000000u);
}
__device__ __forceinline__ float ord2f(unsigned u) {
  return (u & 0x80000000u) ? __uint_as_float(u & 0x7FFFFFFFu)
                           : __uint_as_float(~u);
}

__global__ void init_gmax_kernel(unsigned* __restrict__ gmax, int n)
{
  int i = blockIdx.x * blockDim.x + threadIdx.x;
  if (i < n) gmax[i] = f2ord(-3.402823466e38f);
}

__global__ void gate_max_kernel(const float* __restrict__ gate, const int* __restrict__ batch,
                                unsigned* __restrict__ gmax, int rows)
{
  int n = blockIdx.x * blockDim.x + threadIdx.x;
  if (n < rows) atomicMax(&gmax[batch[n]], f2ord(gate[n]));
}

__global__ void gate_exp_kernel(float* __restrict__ gate, const int* __restrict__ batch,
                                const unsigned* __restrict__ gmax, float* __restrict__ z, int rows)
{
  int n = blockIdx.x * blockDim.x + threadIdx.x;
  if (n >= rows) return;
  int g = batch[n];
  float ex = expf(gate[n] - ord2f(gmax[g]));
  gate[n] = ex;
  atomicAdd(&z[g], ex);
}

// emb[g] = [ att(128) | add(128) ]
__global__ void pool_kernel(const float* __restrict__ x, const float* __restrict__ ex,
                            const float* __restrict__ z, const int* __restrict__ batch,
                            float* __restrict__ emb, int rows)
{
  int t = blockIdx.x * blockDim.x + threadIdx.x;
  if (t >= rows * (HDIM / 4)) return;
  int node = t >> 5;
  int f = (t & 31) << 2;
  int g = batch[node];
  float w = ex[node] / z[g];
  float4 xv = *(const float4*)(x + (size_t)node * HDIM + f);
  float* att = emb + (size_t)g * (2 * HDIM) + f;
  float* add = att + HDIM;
  atomicAdd(att + 0, w * xv.x); atomicAdd(att + 1, w * xv.y);
  atomicAdd(att + 2, w * xv.z); atomicAdd(att + 3, w * xv.w);
  atomicAdd(add + 0, xv.x);     atomicAdd(add + 1, xv.y);
  atomicAdd(add + 2, xv.z);     atomicAdd(add + 3, xv.w);
}

// out[r] = A[r,:K] . w + b[0]   (K multiple of 4)
__global__ void rowdot_kernel(const float* __restrict__ A, const float* __restrict__ w,
                              const float* __restrict__ b, float* __restrict__ out,
                              int rows, int K)
{
  int r = blockIdx.x * blockDim.x + threadIdx.x;
  if (r >= rows) return;
  const float* ap = A + (size_t)r * K;
  float s = 0.f;
  for (int k = 0; k < K; k += 4) {
    float4 v  = *(const float4*)(ap + k);
    float4 ww = *(const float4*)(w + k);
    s += v.x * ww.x + v.y * ww.y + v.z * ww.z + v.w * ww.w;
  }
  out[r] = s + b[0];
}

// ============================================================================
// Host side
// ============================================================================
static inline void launch_gemm(const float* A, const float* A2, const float* W,
                               const float* bias, const float* res, float* Out,
                               int rows, int K, int co, int epi, hipStream_t s)
{
  const int colTiles = co / 16;
  const int rowBlks  = (rows + 63) / 64;           // 4 row-tiles per block
  const int blocks   = rowBlks * colTiles;
  if (A2) wmma_gemm_t<true ><<<blocks, 128, 0, s>>>(A, A2, W, bias, res, Out, rows, K, co, epi);
  else    wmma_gemm_t<false><<<blocks, 128, 0, s>>>(A, A2, W, bias, res, Out, rows, K, co, epi);
}

extern "C" void kernel_launch(void* const* d_in, const int* in_sizes, int n_in,
                              void* d_out, int out_size, void* d_ws, size_t ws_size,
                              hipStream_t stream)
{
  (void)n_in; (void)out_size; (void)ws_size;

  const float* x_in   = (const float*)d_in[0];
  const int*   eidx   = (const int*)d_in[1];
  const float* eattr  = (const float*)d_in[2];
  const int*   batch  = (const int*)d_in[3];
  const float* node_w = (const float*)d_in[4];
  const float* node_b = (const float*)d_in[5];
  const float* edge_w = (const float*)d_in[6];
  const float* edge_b = (const float*)d_in[7];
  const float* gate_w1 = (const float*)d_in[40];
  const float* gate_b1 = (const float*)d_in[41];
  const float* gate_w2 = (const float*)d_in[42];
  const float* gate_b2 = (const float*)d_in[43];
  const float* cl_w1 = (const float*)d_in[44];
  const float* cl_b1 = (const float*)d_in[45];
  const float* cl_w2 = (const float*)d_in[46];
  const float* cl_b2 = (const float*)d_in[47];
  const float* cl_w3 = (const float*)d_in[48];
  const float* cl_b3 = (const float*)d_in[49];

  const int Nn = in_sizes[0] / 32;      // 100000
  const int E_ = in_sizes[1] / 2;       // 400000
  const int* src = eidx;
  const int* dst = eidx + E_;

  // ---- workspace arena -----------------------------------------------------
  char* base = (char*)d_ws;
  size_t off = 0;
  auto alloc = [&](size_t bytes) -> char* {
    char* p = base + off;
    off = (off + bytes + 255) & ~(size_t)255;
    return p;
  };
  float* X    = (float*)alloc((size_t)Nn * HDIM * 4);  //  51.2 MB
  float* AGG  = (float*)alloc((size_t)Nn * HDIM * 4);  //  51.2 MB
  float* Ee   = (float*)alloc((size_t)E_ * HDIM * 4);  // 204.8 MB
  float* H1   = (float*)alloc((size_t)Nn * 256 * 4);   // 102.4 MB (multi-use)
  float* EWP  = (float*)alloc(32 * HDIM * 4);          // padded edge_w (32x128)
  float* SUM  = (float*)alloc(256 * 4);
  float* SQ   = (float*)alloc(256 * 4);
  float* GS   = (float*)alloc(GNUM * 4);
  float* GQ   = (float*)alloc(GNUM * 4);
  float* CNT  = (float*)alloc(GNUM * 4);
  float* Zb   = (float*)alloc(GNUM * 4);
  unsigned* GMAX = (unsigned*)alloc(GNUM * 4);
  // aliases inside H1 region at times it is dead:
  float* EPAD  = H1;                                   // E x 32  (pre-layer-0 only)
  float* GATE1 = H1;                                   // N x 64  (post conv layers)
  float* GATEV = (float*)((char*)H1 + (size_t)Nn * 64 * 4);
  float* EMB   = GATEV + Nn;                           // G x 256
  float* CLH1  = EMB + (size_t)GNUM * 256;             // G x 128
  float* CLH2  = CLH1 + (size_t)GNUM * 128;            // G x 64

  // ---- per-graph node counts (batch is constant across layers) ------------
  hipMemsetAsync(CNT, 0, GNUM * 4, stream);
  count_nodes_kernel<<<(Nn + 255) / 256, 256, 0, stream>>>(batch, CNT, Nn);

  // ---- encoders ------------------------------------------------------------
  launch_gemm(x_in, nullptr, node_w, node_b, nullptr, X, Nn, 32, HDIM, 0, stream);
  // edge encoder has K=16: zero-pad activations (16->32) and weights (16->32
  // rows) once, then run the branch-free K%32==0 WMMA path with K=32.
  pad_rows_kernel<<<(int)(((long)E_ * 32 + 255) / 256), 256, 0, stream>>>(eattr, EPAD, E_, 16, 32);
  pad_rows_kernel<<<(32 * HDIM + 255) / 256, 256, 0, stream>>>(edge_w, EWP, 32, HDIM, HDIM); // copy 16 rows...
  // (edge_w is 16x128 row-major; padding rows means: first 16*128 elems copy,
  //  next 16*128 elems zero -> express as pad over columns of the transposed
  //  view is wrong; do it directly:)
  hipMemsetAsync(EWP + 16 * HDIM, 0, 16 * HDIM * 4, stream);
  hipMemcpyAsync(EWP, edge_w, 16 * HDIM * 4, hipMemcpyDeviceToDevice, stream);
  launch_gemm(EPAD, nullptr, EWP, edge_b, nullptr, Ee, E_, 32, HDIM, 0, stream);

  // ---- 4 GINE layers -------------------------------------------------------
  const int widths[4] = {256, 256, 128, 128};
  for (int i = 0; i < 4; ++i) {
    const float* w1 = (const float*)d_in[8 + 8 * i + 0];
    const float* b1 = (const float*)d_in[8 + 8 * i + 1];
    const float* bg = (const float*)d_in[8 + 8 * i + 2];
    const float* bb = (const float*)d_in[8 + 8 * i + 3];
    const float* w2 = (const float*)d_in[8 + 8 * i + 4];
    const float* b2 = (const float*)d_in[8 + 8 * i + 5];
    const float* lg = (const float*)d_in[8 + 8 * i + 6];
    const float* lb = (const float*)d_in[8 + 8 * i + 7];
    const int w = widths[i];

    hipMemsetAsync(AGG, 0, (size_t)Nn * HDIM * 4, stream);
    msg_agg_kernel<<<(E_ * 32 + 255) / 256, 256, 0, stream>>>(X, Ee, src, dst, AGG, E_);

    // h1 = (x + agg) @ w1 + b1        (A2 fuses the residual-message add)
    launch_gemm(X, AGG, w1, b1, nullptr, H1, Nn, HDIM, w, 0, stream);

    hipMemsetAsync(SUM, 0, 256 * 4, stream);
    hipMemsetAsync(SQ,  0, 256 * 4, stream);
    col_stats_kernel<<<(Nn + 127) / 128, w, 0, stream>>>(H1, SUM, SQ, Nn, w, 128);
    bn_relu_kernel<<<(int)(((long)Nn * w + 255) / 256), 256, 0, stream>>>(H1, SUM, SQ, bg, bb, Nn, w);

    // x = relu(x + relu(h1 @ w2 + b2))   (epi 2, in-place on X)
    launch_gemm(H1, nullptr, w2, b2, X, X, Nn, w, HDIM, 2, stream);

    hipMemsetAsync(GS, 0, GNUM * 4, stream);
    hipMemsetAsync(GQ, 0, GNUM * 4, stream);
    ln_stats_kernel<<<(Nn + 7) / 8, 256, 0, stream>>>(X, batch, GS, GQ, Nn);
    ln_apply_kernel<<<(int)(((long)Nn * HDIM + 255) / 256), 256, 0, stream>>>(
        X, batch, GS, GQ, CNT, lg, lb, Nn);
  }

  // ---- gated softmax pooling ----------------------------------------------
  launch_gemm(X, nullptr, gate_w1, gate_b1, nullptr, GATE1, Nn, HDIM, 64, 1, stream);
  rowdot_kernel<<<(Nn + 255) / 256, 256, 0, stream>>>(GATE1, gate_w2, gate_b2, GATEV, Nn, 64);

  init_gmax_kernel<<<(GNUM + 255) / 256, 256, 0, stream>>>(GMAX, GNUM);
  hipMemsetAsync(Zb, 0, GNUM * 4, stream);
  gate_max_kernel<<<(Nn + 255) / 256, 256, 0, stream>>>(GATEV, batch, GMAX, Nn);
  gate_exp_kernel<<<(Nn + 255) / 256, 256, 0, stream>>>(GATEV, batch, GMAX, Zb, Nn);

  hipMemsetAsync(EMB, 0, (size_t)GNUM * 256 * 4, stream);
  pool_kernel<<<(Nn * 32 + 255) / 256, 256, 0, stream>>>(X, GATEV, Zb, batch, EMB, Nn);

  // ---- classifier ----------------------------------------------------------
  launch_gemm(EMB,  nullptr, cl_w1, cl_b1, nullptr, CLH1, GNUM, 256, HDIM, 1, stream);
  launch_gemm(CLH1, nullptr, cl_w2, cl_b2, nullptr, CLH2, GNUM, HDIM, 64, 1, stream);
  rowdot_kernel<<<(GNUM + 255) / 256, 256, 0, stream>>>(CLH2, cl_w3, cl_b3, (float*)d_out, GNUM, 64);
}
